// Decoder_2422361555423
// MI455X (gfx1250) — matrix operand order repaired and checked by reading, not compiled, not run on hardware
//
#include <hip/hip_runtime.h>
#include <math.h>

typedef _Float16 h16;
typedef __attribute__((ext_vector_type(16))) _Float16     v16h;
typedef __attribute__((ext_vector_type(8)))  float        v8f;
typedef __attribute__((ext_vector_type(4)))  unsigned int u32x4;
typedef __attribute__((ext_vector_type(4)))  float        f32x4;

union FragH { v16h v; u32x4 q[2]; h16 h[16]; };

#define GF_GELU   1
#define GF_RESF32 2
#define GF_TRANS  4

// ---------------------------------------------------------------- transposes
__global__ void k_transpose_in(const float* __restrict__ x, float* __restrict__ X) {
  // x: [8][675][30][256] -> X: [(b0*30+t)*675+n][256], vectorized by float4
  size_t i = (size_t)blockIdx.x * blockDim.x + threadIdx.x;
  const size_t total = (size_t)8 * 675 * 30 * 64;
  if (i >= total) return;
  int c4 = (int)(i % 64); size_t r = i / 64;
  int t = (int)(r % 30); r /= 30;
  int n = (int)(r % 675); int b0 = (int)(r / 675);
  f32x4 v = ((const f32x4*)x)[i];
  size_t dst = ((size_t)(b0 * 30 + t) * 675 + n) * 64 + c4;
  ((f32x4*)X)[dst] = v;
}

__global__ void k_transpose_out(const float* __restrict__ X, float* __restrict__ o) {
  size_t i = (size_t)blockIdx.x * blockDim.x + threadIdx.x;
  const size_t total = (size_t)8 * 675 * 30 * 64;
  if (i >= total) return;
  int c4 = (int)(i % 64); size_t r = i / 64;
  int t = (int)(r % 30); r /= 30;
  int n = (int)(r % 675); int b0 = (int)(r / 675);
  size_t src = ((size_t)(b0 * 30 + t) * 675 + n) * 64 + c4;
  ((f32x4*)o)[i] = ((const f32x4*)X)[src];
}

// ---------------------------------------------------------------- conversions
__global__ void k_f32f16(const float* __restrict__ s, h16* __restrict__ d, size_t n) {
  size_t i = (size_t)blockIdx.x * blockDim.x + threadIdx.x;
  if (i < n) d[i] = (h16)s[i];
}

__global__ void k_wT(const float* __restrict__ w, h16* __restrict__ wt, int K, int N) {
  // w: [K][N] f32 -> wt: [N][K] f16
  size_t i = (size_t)blockIdx.x * blockDim.x + threadIdx.x;
  if (i >= (size_t)K * N) return;
  int n = (int)(i % N); int k = (int)(i / N);
  wt[(size_t)n * K + k] = (h16)w[i];
}

__global__ void k_w3d(const float* __restrict__ w3, float* __restrict__ d) {
  int k = threadIdx.x;
  if (k < 64) d[k] = w3[2 * k] - w3[2 * k + 1];
}

__global__ void k_fill(unsigned int* __restrict__ p, size_t n) {
  size_t i = (size_t)blockIdx.x * blockDim.x + threadIdx.x;
  if (i < n) p[i] = 0u;
}

// ---------------------------------------------------------------- layernorm (wave per row, optional gather)
__global__ __launch_bounds__(256) void k_ln(
    const float* __restrict__ X, const float* __restrict__ g, const float* __restrict__ bta,
    h16* __restrict__ out, int rows, const int* __restrict__ gidx, int nFull, int nkOut) {
  int wid = blockIdx.x * 8 + (threadIdx.x >> 5);
  if (wid >= rows) return;
  int lane = threadIdx.x & 31;
  size_t srow;
  if (gidx) { int bb = wid / nkOut; int j = wid - bb * nkOut;
              srow = (size_t)bb * nFull + gidx[bb * nkOut + j]; }
  else srow = (size_t)wid;
  const f32x4* p = (const f32x4*)(X + srow * 256);
  f32x4 a = p[lane * 2], c = p[lane * 2 + 1];
  float s  = a.x + a.y + a.z + a.w + c.x + c.y + c.z + c.w;
  float s2 = a.x*a.x + a.y*a.y + a.z*a.z + a.w*a.w + c.x*c.x + c.y*c.y + c.z*c.z + c.w*c.w;
#pragma unroll
  for (int d = 1; d < 32; d <<= 1) { s += __shfl_xor(s, d, 32); s2 += __shfl_xor(s2, d, 32); }
  float mu  = s * (1.f / 256.f);
  float inv = rsqrtf(s2 * (1.f / 256.f) - mu * mu + 1e-5f);
  float vals[8] = {a.x, a.y, a.z, a.w, c.x, c.y, c.z, c.w};
  h16* o = out + (size_t)wid * 256 + lane * 8;
  const float* gg = g + lane * 8;
  const float* bb2 = bta + lane * 8;
#pragma unroll
  for (int i = 0; i < 8; ++i) o[i] = (h16)((vals[i] - mu) * inv * gg[i] + bb2[i]);
}

// ---------------------------------------------------------------- WMMA GEMM (wave = 16x64 tile)
// Double-buffered k-loop: requires K % 64 == 0 (all call sites use K in {128, 256}).
__global__ __launch_bounds__(256) void k_gemm(
    const h16* __restrict__ A, int lda,
    const h16* __restrict__ Bw, int ldb,     // Bw: [N][K] f16 (transposed weights)
    const float* __restrict__ bias,
    void* __restrict__ out,
    int M, int N, int K, int flags, int rowsPerB, int ldT) {
  int lane = threadIdx.x & 31;
  int wid  = blockIdx.x * 8 + (threadIdx.x >> 5);
  int mTiles = M >> 4, nT4 = N >> 6;
  if (wid >= mTiles * nT4) return;
  int m0 = (wid % mTiles) << 4;
  int n0 = (wid / mTiles) << 6;
  int l15 = lane & 15;
  int hi  = lane >> 4;
  v8f acc[4];
#pragma unroll
  for (int t = 0; t < 4; ++t)
#pragma unroll
    for (int j = 0; j < 8; ++j) acc[t][j] = 0.f;

  const h16* ap  = A  + (size_t)(m0 + l15) * lda + (hi ? 8 : 0);
  const h16* bp0 = Bw + (size_t)(n0 + l15) * ldb + (hi ? 8 : 0);
  const size_t bstep = (size_t)16 * ldb;

  FragH a0, a1, b0[4], b1[4];
  // prologue: stage k = 0
  a0.q[0] = *(const u32x4*)(ap);
  a0.q[1] = *(const u32x4*)(ap + 16);
#pragma unroll
  for (int t = 0; t < 4; ++t) {
    const h16* bp = bp0 + (size_t)t * bstep;
    b0[t].q[0] = *(const u32x4*)(bp);
    b0[t].q[1] = *(const u32x4*)(bp + 16);
  }
  int k0 = 0;
  for (; k0 < K - 64; k0 += 64) {
    // stage k0+32 while computing k0
    a1.q[0] = *(const u32x4*)(ap + k0 + 32);
    a1.q[1] = *(const u32x4*)(ap + k0 + 48);
#pragma unroll
    for (int t = 0; t < 4; ++t) {
      const h16* bp = bp0 + (size_t)t * bstep + k0 + 32;
      b1[t].q[0] = *(const u32x4*)(bp);
      b1[t].q[1] = *(const u32x4*)(bp + 16);
    }
#pragma unroll
    for (int t = 0; t < 4; ++t)
      acc[t] = __builtin_amdgcn_wmma_f32_16x16x32_f16(false, a0.v, false, b0[t].v,
                                                      (short)0, acc[t], false, false);
    // stage k0+64 while computing k0+32
    a0.q[0] = *(const u32x4*)(ap + k0 + 64);
    a0.q[1] = *(const u32x4*)(ap + k0 + 80);
#pragma unroll
    for (int t = 0; t < 4; ++t) {
      const h16* bp = bp0 + (size_t)t * bstep + k0 + 64;
      b0[t].q[0] = *(const u32x4*)(bp);
      b0[t].q[1] = *(const u32x4*)(bp + 16);
    }
#pragma unroll
    for (int t = 0; t < 4; ++t)
      acc[t] = __builtin_amdgcn_wmma_f32_16x16x32_f16(false, a1.v, false, b1[t].v,
                                                      (short)0, acc[t], false, false);
  }
  // epilogue: last 64 of K (stage k0+32 while computing k0)
  a1.q[0] = *(const u32x4*)(ap + k0 + 32);
  a1.q[1] = *(const u32x4*)(ap + k0 + 48);
#pragma unroll
  for (int t = 0; t < 4; ++t) {
    const h16* bp = bp0 + (size_t)t * bstep + k0 + 32;
    b1[t].q[0] = *(const u32x4*)(bp);
    b1[t].q[1] = *(const u32x4*)(bp + 16);
  }
#pragma unroll
  for (int t = 0; t < 4; ++t)
    acc[t] = __builtin_amdgcn_wmma_f32_16x16x32_f16(false, a0.v, false, b0[t].v,
                                                    (short)0, acc[t], false, false);
#pragma unroll
  for (int t = 0; t < 4; ++t)
    acc[t] = __builtin_amdgcn_wmma_f32_16x16x32_f16(false, a1.v, false, b1[t].v,
                                                    (short)0, acc[t], false, false);

  int rofs = hi ? 8 : 0;
#pragma unroll
  for (int t = 0; t < 4; ++t) {
    int n = n0 + t * 16 + l15;
    float bv = bias ? bias[n] : 0.f;
#pragma unroll
    for (int j = 0; j < 8; ++j) {
      int m = m0 + j + rofs;
      float v = acc[t][j] + bv;
      if (flags & GF_GELU) v = 0.5f * v * (1.f + erff(v * 0.70710678118f));
      if (flags & GF_RESF32) {
        ((float*)out)[(size_t)m * N + n] += v;
      } else if (flags & GF_TRANS) {
        int bb = m / rowsPerB;
        int kk = m - bb * rowsPerB;
        ((h16*)out)[((size_t)bb * N + n) * ldT + kk] = (h16)v;
      } else {
        ((h16*)out)[(size_t)m * N + n] = (h16)v;
      }
    }
  }
}

// ---------------------------------------------------------------- flash attention (wave per 16-query tile)
__global__ __launch_bounds__(32) void k_flash(
    const h16* __restrict__ Q, const h16* __restrict__ Kd, const h16* __restrict__ Vt,
    h16* __restrict__ O, int Nq, int Nk, int ldVt, float scale, int qTiles) {
  __shared__ h16 Pld[16 * 32];
  int b  = blockIdx.x / qTiles;
  int qt = blockIdx.x % qTiles;
  int lane = threadIdx.x & 31;
  int l15 = lane & 15;
  int hi  = lane >> 4;
  int m0 = qt * 16;
  int qrow = m0 + l15; if (qrow >= Nq) qrow = Nq - 1;

  FragH qf[8];
  {
    const h16* qp = Q + ((size_t)b * Nq + qrow) * 256 + (hi ? 8 : 0);
#pragma unroll
    for (int i = 0; i < 8; ++i) {
      qf[i].q[0] = *(const u32x4*)(qp + i * 32);
      qf[i].q[1] = *(const u32x4*)(qp + i * 32 + 16);
    }
  }
  v8f o[16];
#pragma unroll
  for (int c = 0; c < 16; ++c)
#pragma unroll
    for (int j = 0; j < 8; ++j) o[c][j] = 0.f;
  float mrun[8], lrun[8], alpha[8];
#pragma unroll
  for (int j = 0; j < 8; ++j) { mrun[j] = -3.0e38f; lrun[j] = 0.f; }

  for (int kt = 0; kt < Nk; kt += 32) {
    int keyA = kt + l15, keyB = keyA + 16;
    int kcA = (keyA < Nk) ? keyA : Nk - 1;
    int kcB = (keyB < Nk) ? keyB : Nk - 1;
    const h16* kpA = Kd + ((size_t)b * Nk + kcA) * 256 + (hi ? 8 : 0);
    const h16* kpB = Kd + ((size_t)b * Nk + kcB) * 256 + (hi ? 8 : 0);
    v8f sa, sb;
#pragma unroll
    for (int j = 0; j < 8; ++j) { sa[j] = 0.f; sb[j] = 0.f; }
#pragma unroll
    for (int i = 0; i < 8; ++i) {
      FragH ka, kb;
      ka.q[0] = *(const u32x4*)(kpA + i * 32); ka.q[1] = *(const u32x4*)(kpA + i * 32 + 16);
      kb.q[0] = *(const u32x4*)(kpB + i * 32); kb.q[1] = *(const u32x4*)(kpB + i * 32 + 16);
      sa = __builtin_amdgcn_wmma_f32_16x16x32_f16(false, qf[i].v, false, ka.v, (short)0, sa, false, false);
      sb = __builtin_amdgcn_wmma_f32_16x16x32_f16(false, qf[i].v, false, kb.v, (short)0, sb, false, false);
    }
#pragma unroll
    for (int j = 0; j < 8; ++j) {
      float va = (keyA < Nk) ? sa[j] * scale : -3.0e38f;
      float vb = (keyB < Nk) ? sb[j] * scale : -3.0e38f;
      float mx = fmaxf(va, vb);
#pragma unroll
      for (int d = 1; d < 16; d <<= 1) mx = fmaxf(mx, __shfl_xor(mx, d, 32));
      float mn = fmaxf(mrun[j], mx);
      float al = __expf(mrun[j] - mn);
      float pa = __expf(va - mn);
      float pb = __expf(vb - mn);
      float rs = pa + pb;
#pragma unroll
      for (int d = 1; d < 16; d <<= 1) rs += __shfl_xor(rs, d, 32);
      lrun[j] = lrun[j] * al + rs;
      mrun[j] = mn;
      alpha[j] = al;
      int prow = j + (hi ? 8 : 0);
      Pld[prow * 32 + l15]      = (h16)pa;
      Pld[prow * 32 + l15 + 16] = (h16)pb;
    }
#pragma unroll
    for (int c = 0; c < 16; ++c)
#pragma unroll
      for (int j = 0; j < 8; ++j) o[c][j] *= alpha[j];
    FragH pf;
    {
      const h16* pp = Pld + l15 * 32 + (hi ? 8 : 0);
      pf.q[0] = *(const u32x4*)(pp);
      pf.q[1] = *(const u32x4*)(pp + 16);
    }
#pragma unroll
    for (int c = 0; c < 16; ++c) {
      const h16* vp = Vt + ((size_t)b * 256 + c * 16 + l15) * ldVt + kt + (hi ? 8 : 0);
      FragH vf;
      vf.q[0] = *(const u32x4*)(vp);
      vf.q[1] = *(const u32x4*)(vp + 16);
      o[c] = __builtin_amdgcn_wmma_f32_16x16x32_f16(false, pf.v, false, vf.v, (short)0, o[c], false, false);
    }
  }
#pragma unroll
  for (int j = 0; j < 8; ++j) lrun[j] = 1.f / lrun[j];
#pragma unroll
  for (int c = 0; c < 16; ++c)
#pragma unroll
    for (int j = 0; j < 8; ++j) {
      int m = m0 + j + (hi ? 8 : 0);
      if (m < Nq)
        O[((size_t)b * Nq + m) * 256 + c * 16 + l15] = (h16)(o[c][j] * lrun[j]);
    }
}

// ---------------------------------------------------------------- predictor helpers
__global__ void k_gmean(const h16* __restrict__ Ph, float* __restrict__ g, int Nn) {
  int b = blockIdx.x, c = threadIdx.x;      // grid 240 x 128
  const h16* p = Ph + (size_t)b * Nn * 256 + 128 + c;
  float s = 0.f;
  for (int n = 0; n < Nn; ++n) s += (float)p[(size_t)n * 256];
  g[b * 128 + c] = s / (float)Nn;
}

__global__ void k_concat(const h16* __restrict__ Ph, const float* __restrict__ g,
                         h16* __restrict__ out, int Nn) {
  size_t i = (size_t)blockIdx.x * blockDim.x + threadIdx.x;
  const size_t total = (size_t)240 * 675 * 256;
  if (i >= total) return;
  int c = (int)(i & 255); size_t r = i >> 8;
  int b = (int)(r / Nn);
  out[i] = (c < 128) ? Ph[i] : (h16)g[b * 128 + (c - 128)];
}

__global__ void k_score(const h16* __restrict__ h3, const float* __restrict__ w3d,
                        float* __restrict__ sc, int rows) {
  size_t r = (size_t)blockIdx.x * blockDim.x + threadIdx.x;
  if (r >= (size_t)rows) return;
  const h16* p = h3 + r * 64;
  float s = 0.f;
#pragma unroll
  for (int k = 0; k < 64; ++k) s += (float)p[k] * w3d[k];
  sc[r] = s;
}

__global__ void k_topk(const float* __restrict__ sc, int* __restrict__ keep,
                       int Nn, int nkeep) {
  __shared__ float s[675];
  int b = blockIdx.x;
  for (int i = threadIdx.x; i < Nn; i += blockDim.x) s[i] = sc[(size_t)b * Nn + i];
  __syncthreads();
  int t = threadIdx.x;
  if (t < Nn) {
    float st = s[t];
    int rank = 0;
    for (int j = 0; j < Nn; ++j) {
      float sj = s[j];
      rank += (sj > st) || (sj == st && j < t);
    }
    if (rank < nkeep) keep[b * nkeep + rank] = t;
  }
}

// ---------------------------------------------------------------- driver
extern "C" void kernel_launch(void* const* d_in, const int* in_sizes, int n_in,
                              void* d_out, int out_size, void* d_ws, size_t ws_size,
                              hipStream_t stream) {
  (void)in_sizes; (void)out_size;
  if (n_in < 32) return;
  const float* x_in    = (const float*)d_in[0];
  const float* mem_in  = (const float*)d_in[1];
  const float* p_ln_g  = (const float*)d_in[4];
  const float* p_ln_b  = (const float*)d_in[5];
  const float* p_in_w  = (const float*)d_in[6];
  const float* p_in_b  = (const float*)d_in[7];
  const float* p_w1    = (const float*)d_in[8];
  const float* p_b1    = (const float*)d_in[9];
  const float* p_w2    = (const float*)d_in[10];
  const float* p_b2    = (const float*)d_in[11];
  const float* p_w3    = (const float*)d_in[12];
  const float* l_ln1_g = (const float*)d_in[14];
  const float* l_ln1_b = (const float*)d_in[15];
  const float* l_wq    = (const float*)d_in[16];
  const float* l_wk    = (const float*)d_in[17];
  const float* l_wv    = (const float*)d_in[18];
  const float* l_wo    = (const float*)d_in[19];
  const float* l_ln2_g = (const float*)d_in[20];
  const float* l_ln2_b = (const float*)d_in[21];
  const float* l_cq    = (const float*)d_in[22];
  const float* l_ck    = (const float*)d_in[23];
  const float* l_cv    = (const float*)d_in[24];
  const float* l_co    = (const float*)d_in[25];
  const float* l_ln3_g = (const float*)d_in[26];
  const float* l_ln3_b = (const float*)d_in[27];
  const float* l_w1    = (const float*)d_in[28];
  const float* l_fb1   = (const float*)d_in[29];
  const float* l_w2    = (const float*)d_in[30];
  const float* l_fb2   = (const float*)d_in[31];

  const int ROWS = 240 * 675;          // 162000
  const size_t BIGH = (size_t)240 * 256 * 704;

  char* base = (char*)d_ws;
  size_t off = 0;
  auto alloc = [&](size_t bytes) -> char* {
    char* p = base + off;
    off = (off + bytes + 255) & ~(size_t)255;
    return p;
  };
  float* X   = (float*)alloc((size_t)ROWS * 256 * 4);
  h16* f1    = (h16*)alloc(BIGH * 2);
  h16* f2    = (h16*)alloc(BIGH * 2);
  h16* f3    = (h16*)alloc(BIGH * 2);
  h16* f4    = (h16*)alloc(BIGH * 2);      // Vt slot (padded ld)
  h16* f5    = (h16*)alloc(BIGH * 2);
  h16* f6    = (h16*)alloc((size_t)240 * 540 * 256 * 2);
  h16* memh  = (h16*)alloc((size_t)30720 * 256 * 2);
  h16* MK    = (h16*)alloc((size_t)30720 * 256 * 2);
  h16* MVt   = (h16*)alloc((size_t)240 * 256 * 128 * 2);
  h16* wqT   = (h16*)alloc(65536 * 2);
  h16* wkT   = (h16*)alloc(65536 * 2);
  h16* wvT   = (h16*)alloc(65536 * 2);
  h16* woT   = (h16*)alloc(65536 * 2);
  h16* cqT   = (h16*)alloc(65536 * 2);
  h16* ckT   = (h16*)alloc(65536 * 2);
  h16* cvT   = (h16*)alloc(65536 * 2);
  h16* coT   = (h16*)alloc(65536 * 2);
  h16* w1T   = (h16*)alloc((size_t)1024 * 256 * 2);
  h16* w2T   = (h16*)alloc((size_t)256 * 1024 * 2);
  h16* pinT  = (h16*)alloc(65536 * 2);
  h16* pw1T  = (h16*)alloc((size_t)128 * 256 * 2);
  h16* pw2T  = (h16*)alloc((size_t)64 * 128 * 2);
  float* w3d   = (float*)alloc(64 * 4);
  float* gmean = (float*)alloc((size_t)240 * 128 * 4);
  float* scores = (float*)alloc((size_t)ROWS * 4);
  int* keep    = (int*)alloc((size_t)240 * 540 * 4);
  if (off > ws_size) return;

  auto gemm = [&](const h16* A, int lda, const h16* Bw, int ldb, const float* bias,
                  void* out, int M, int N, int K, int flags, int rpb, int ldT) {
    int tiles = (M >> 4) * (N >> 6);
    k_gemm<<<(tiles + 7) / 8, 256, 0, stream>>>(A, lda, Bw, ldb, bias, out, M, N, K, flags, rpb, ldT);
  };
  auto ln = [&](const float* src, const float* g, const float* b, h16* o, int rows,
                const int* gi, int nfull, int nko) {
    k_ln<<<(rows + 7) / 8, 256, 0, stream>>>(src, g, b, o, rows, gi, nfull, nko);
  };
  auto wT = [&](const float* w, h16* wt, int K, int N) {
    size_t n = (size_t)K * N;
    k_wT<<<(unsigned)((n + 255) / 256), 256, 0, stream>>>(w, wt, K, N);
  };

  // ---- one-time prep (per call) ----
  {
    size_t tot4 = (size_t)8 * 675 * 30 * 64;
    k_transpose_in<<<(unsigned)((tot4 + 255) / 256), 256, 0, stream>>>(x_in, X);
  }
  wT(l_wq, wqT, 256, 256);  wT(l_wk, wkT, 256, 256);
  wT(l_wv, wvT, 256, 256);  wT(l_wo, woT, 256, 256);
  wT(l_cq, cqT, 256, 256);  wT(l_ck, ckT, 256, 256);
  wT(l_cv, cvT, 256, 256);  wT(l_co, coT, 256, 256);
  wT(l_w1, w1T, 256, 1024); wT(l_w2, w2T, 1024, 256);
  wT(p_in_w, pinT, 256, 256); wT(p_w1, pw1T, 256, 128); wT(p_w2, pw2T, 128, 64);
  k_w3d<<<1, 64, 0, stream>>>(p_w3, w3d);
  {
    size_t n = (size_t)30720 * 256;
    k_f32f16<<<(unsigned)((n + 255) / 256), 256, 0, stream>>>(mem_in, memh, n);
  }
  { // zero Vt slot so padded key columns stay finite
    size_t n = BIGH / 2;
    k_fill<<<(unsigned)((n + 255) / 256), 256, 0, stream>>>((unsigned int*)f4, n);
  }
  // cross-attention K/V from memory (shared by all layers: same params)
  gemm(memh, 256, ckT, 256, nullptr, MK, 30720, 256, 256, 0, 0, 0);
  gemm(memh, 256, cvT, 256, nullptr, MVt, 30720, 256, 256, GF_TRANS, 128, 128);

  const float scale = 0.0625f;   // 1/sqrt(256)
  const int QT = 43;             // ceil(675/16)

  for (int layer = 0; layer < 3; ++layer) {
    if (layer == 1) {
      // ---- predictor + top-k selection ----
      ln(X, p_ln_g, p_ln_b, f1, ROWS, nullptr, 0, 0);
      gemm(f1, 256, pinT, 256, p_in_b, f2, ROWS, 256, 256, GF_GELU, 0, 0);
      k_gmean<<<240, 128, 0, stream>>>(f2, gmean, 675);
      {
        size_t n = (size_t)ROWS * 256;
        k_concat<<<(unsigned)((n + 255) / 256), 256, 0, stream>>>(f2, gmean, f3, 675);
      }
      gemm(f3, 256, pw1T, 256, p_b1, f4, ROWS, 128, 256, GF_GELU, 0, 0);
      gemm(f4, 128, pw2T, 128, p_b2, f5, ROWS, 64, 128, GF_GELU, 0, 0);
      k_score<<<(ROWS + 255) / 256, 256, 0, stream>>>(f5, w3d, scores, ROWS);
      k_topk<<<240, 704, 0, stream>>>(scores, keep, 675, 540);
    }
    int Nk, ldV;
    if (layer == 1) { Nk = 540; ldV = 544; } else { Nk = 675; ldV = 704; }
    int Mk = 240 * Nk;

    // ---- self attention ----
    ln(X, l_ln1_g, l_ln1_b, f1, ROWS, nullptr, 0, 0);
    h16* Hk = f1;
    if (layer == 1) { ln(X, l_ln1_g, l_ln1_b, f6, Mk, keep, 675, 540); Hk = f6; }
    gemm(f1, 256, wqT, 256, nullptr, f2, ROWS, 256, 256, 0, 0, 0);
    gemm(Hk, 256, wkT, 256, nullptr, f3, Mk, 256, 256, 0, 0, 0);
    gemm(Hk, 256, wvT, 256, nullptr, f4, Mk, 256, 256, GF_TRANS, Nk, ldV);
    k_flash<<<240 * QT, 32, 0, stream>>>(f2, f3, f4, f5, 675, Nk, ldV, scale, QT);
    gemm(f5, 256, woT, 256, nullptr, X, ROWS, 256, 256, GF_RESF32, 0, 0);

    // ---- cross attention ----
    ln(X, l_ln2_g, l_ln2_b, f1, ROWS, nullptr, 0, 0);
    gemm(f1, 256, cqT, 256, nullptr, f2, ROWS, 256, 256, 0, 0, 0);
    k_flash<<<240 * QT, 32, 0, stream>>>(f2, MK, MVt, f5, 675, 128, 128, scale, QT);
    gemm(f5, 256, coT, 256, nullptr, X, ROWS, 256, 256, GF_RESF32, 0, 0);

    // ---- FFN (4 column chunks of 256) ----
    ln(X, l_ln3_g, l_ln3_b, f1, ROWS, nullptr, 0, 0);
    for (int c = 0; c < 4; ++c) {
      gemm(f1, 256, w1T + (size_t)c * 256 * 256, 256, l_fb1 + c * 256,
           f2, ROWS, 256, 256, GF_GELU, 0, 0);
      gemm(f2, 256, w2T + c * 256, 1024, (c == 0) ? l_fb2 : nullptr,
           X, ROWS, 256, 256, GF_RESF32, 0, 0);
    }
  }

  {
    size_t tot4 = (size_t)8 * 675 * 30 * 64;
    k_transpose_out<<<(unsigned)((tot4 + 255) / 256), 256, 0, stream>>>(X, (float*)d_out);
  }
}
